// TemporalAttention_39470749450367
// MI455X (gfx1250) — matrix operand order repaired and checked by reading, not compiled, not run on hardware
//
#include <hip/hip_runtime.h>
#include <hip/hip_bf16.h>
#include <math.h>

// Problem constants (from the reference)
#define F_DIM 256
#define H_DIM 256
#define BATCH 64
#define TIME  4096
#define M_ROWS (BATCH * TIME)          // 262144 flattened (b,t) rows

typedef __attribute__((ext_vector_type(16))) __bf16 v16bf;
typedef __attribute__((ext_vector_type(8)))  float  v8f;

union BF16x16 {
    v16bf v;
    unsigned short u[16];
    uint4 q[2];
};

// f32 -> bf16, round to nearest even
__device__ __forceinline__ unsigned short f2bf(float f) {
    unsigned int x = __float_as_uint(f);
    unsigned int r = x + 0x7FFFu + ((x >> 16) & 1u);
    return (unsigned short)(r >> 16);
}

// Branchless fast tanh: tanh(v) = 1 - 2/(exp(2v)+1).
// exp(2v)->inf gives +1, exp(2v)->0 gives -1; monotone in between.
// One v_exp_f32 (transcendental, co-executes with VALU) + rcp + fma.
__device__ __forceinline__ float fast_tanh(float v) {
    float e2 = __expf(2.0f * v);
    return 1.0f - 2.0f / (e2 + 1.0f);
}

// ---------------------------------------------------------------------------
// Kernel 1: fold Wq+Wk -> bf16, written directly in WMMA B-fragment order.
// Fragment layout for V_WMMA_F32_16X16X32_BF16 B (32x16, KxN):
//   lane l: n = l & 15, khalf = l >> 4; element i (0..7) <-> K = khalf*8 + i, i (8..15) <-> K = 16 + khalf*8 + (i-8)
// Flat index: (((j*8 + kb)*32 + l)*16 + i), j = N-tile (h/16), kb = K-step (f/32)
// Also zeroes the context section of d_out.
// ---------------------------------------------------------------------------
__global__ __launch_bounds__(256) void prepK(const float* __restrict__ Wq,
                                             const float* __restrict__ Wk,
                                             unsigned short* __restrict__ wcf,
                                             float* __restrict__ ctx) {
    int tid = blockIdx.x * 256 + threadIdx.x;      // 0 .. 65535
    int i  = tid & 15;
    int l  = (tid >> 4) & 31;
    int kb = (tid >> 9) & 7;
    int j  = tid >> 12;
    int h  = j * 16 + (l & 15);
    int f  = kb * 32 + ((l >> 4) * 8) + (i & 7) + ((i >> 3) * 16);
    float v = Wq[h * F_DIM + f] + Wk[h * F_DIM + f];
    wcf[tid] = f2bf(v);
    if (tid < BATCH * F_DIM) ctx[tid] = 0.0f;
}

// ---------------------------------------------------------------------------
// Kernel 2: fused scores = tanh(X*Wc^T + bc) . Ws + bs  via bf16 WMMA.
// One wave per 16-row M-tile; A-frags (16x32 bf16) held in registers for all
// 8 K-steps; 16 N-tiles x 8 K-steps of v_wmma_f32_16x16x32_bf16 per wave.
// A layout (16-bit A 16x32): lane l: m = l&15, kgroup = l>>4;
//   elements 0..7  <-> K = kgroup*8 + 0..7
//   elements 8..15 <-> K = 16 + kgroup*8 + 0..7
// C layout: VGPR r, lane l: M = (l>>4)*8 + r, N = l&15.
// ---------------------------------------------------------------------------
__global__ __launch_bounds__(256) void scoresK(const float* __restrict__ X,
                                               const unsigned short* __restrict__ wcf,
                                               const float* __restrict__ bq,
                                               const float* __restrict__ bk,
                                               const float* __restrict__ Ws,
                                               const float* __restrict__ bs,
                                               float* __restrict__ scoresOut) {
    const int lane   = threadIdx.x & 31;
    const int wv     = threadIdx.x >> 5;
    const int mtile  = blockIdx.x * 8 + wv;
    const int m0     = mtile * 16;
    const int mrow   = m0 + (lane & 15);
    const int kgroup = lane >> 4;

    const float* xr = X + (size_t)mrow * F_DIM + kgroup * 8;

    // Load + convert the 16x256 f32 X tile into 8 bf16 A-fragments
    BF16x16 a[8];
#pragma unroll
    for (int kb = 0; kb < 8; ++kb) {
        const float4* p0 = (const float4*)(xr + kb * 32);        // K = kb*32 + kgroup*8 + 0..7
        const float4* p1 = (const float4*)(xr + kb * 32 + 16);   // K = kb*32 + 16 + kgroup*8 + 0..7
        float4 c0 = p0[0], c1 = p0[1], c2 = p1[0], c3 = p1[1];
        a[kb].u[0]  = f2bf(c0.x);  a[kb].u[1]  = f2bf(c0.y);
        a[kb].u[2]  = f2bf(c0.z);  a[kb].u[3]  = f2bf(c0.w);
        a[kb].u[4]  = f2bf(c1.x);  a[kb].u[5]  = f2bf(c1.y);
        a[kb].u[6]  = f2bf(c1.z);  a[kb].u[7]  = f2bf(c1.w);
        a[kb].u[8]  = f2bf(c2.x);  a[kb].u[9]  = f2bf(c2.y);
        a[kb].u[10] = f2bf(c2.z);  a[kb].u[11] = f2bf(c2.w);
        a[kb].u[12] = f2bf(c3.x);  a[kb].u[13] = f2bf(c3.y);
        a[kb].u[14] = f2bf(c3.z);  a[kb].u[15] = f2bf(c3.w);
    }

    float sc[8];
#pragma unroll
    for (int r = 0; r < 8; ++r) sc[r] = 0.0f;

    for (int j = 0; j < 16; ++j) {
        const int h   = j * 16 + (lane & 15);    // hidden index this lane's N maps to
        const float wsv = Ws[h];
        const float bcv = bq[h] + bk[h];

        v8f acc = {0.f, 0.f, 0.f, 0.f, 0.f, 0.f, 0.f, 0.f};
#pragma unroll
        for (int kb = 0; kb < 8; ++kb) {
            BF16x16 bf;
            const uint4* bp = (const uint4*)(wcf + (((size_t)(j * 8 + kb) * 32 + lane) * 16));
            bf.q[0] = bp[0];
            bf.q[1] = bp[1];
            acc = __builtin_amdgcn_wmma_f32_16x16x32_bf16(
                /*neg_a=*/false, a[kb].v, /*neg_b=*/false, bf.v,
                /*c_mod=*/(short)0, acc, /*reuse_a=*/false, /*reuse_b=*/false);
        }
        // Fused epilogue: score += Ws[h] * tanh(hidden + bc[h]) — branchless
#pragma unroll
        for (int r = 0; r < 8; ++r) {
            sc[r] += wsv * fast_tanh(acc[r] + bcv);
        }
    }

    // Reduce across the 16 lanes sharing the same M row (xor within 16-group)
#pragma unroll
    for (int r = 0; r < 8; ++r) {
        float v = sc[r];
        v += __shfl_xor(v, 1, 32);
        v += __shfl_xor(v, 2, 32);
        v += __shfl_xor(v, 4, 32);
        v += __shfl_xor(v, 8, 32);
        sc[r] = v;
    }

    if ((lane & 15) == 0) {
        const float bsv = bs[0];
        const int base = m0 + kgroup * 8;        // lane 0 -> rows m0..m0+7, lane 16 -> m0+8..m0+15
#pragma unroll
        for (int r = 0; r < 8; ++r) scoresOut[base + r] = sc[r] + bsv;
    }
}

// ---------------------------------------------------------------------------
// Kernel 3: in-place softmax over T=4096 per batch row.
// ---------------------------------------------------------------------------
__global__ __launch_bounds__(256) void softmaxK(float* __restrict__ w) {
    __shared__ float red[256];
    float* row = w + (size_t)blockIdx.x * TIME;
    const int tx = threadIdx.x;

    float lmax = -INFINITY;
    for (int t = tx; t < TIME; t += 256) lmax = fmaxf(lmax, row[t]);
    red[tx] = lmax;
    __syncthreads();
    for (int s = 128; s > 0; s >>= 1) {
        if (tx < s) red[tx] = fmaxf(red[tx], red[tx + s]);
        __syncthreads();
    }
    const float mx = red[0];
    __syncthreads();

    float lsum = 0.0f;
    for (int t = tx; t < TIME; t += 256) {
        float e = __expf(row[t] - mx);
        row[t] = e;
        lsum += e;
    }
    red[tx] = lsum;
    __syncthreads();
    for (int s = 128; s > 0; s >>= 1) {
        if (tx < s) red[tx] += red[tx + s];
        __syncthreads();
    }
    const float inv = 1.0f / red[0];
    for (int t = tx; t < TIME; t += 256) row[t] *= inv;
}

// ---------------------------------------------------------------------------
// Kernel 4: context[b,f] = sum_t w[b,t] * x[b,t,f].
// Grid (BATCH, 16): each block handles a 256-long T chunk; coalesced over F;
// f32 atomicAdd accumulation into pre-zeroed d_out.
// ---------------------------------------------------------------------------
__global__ __launch_bounds__(256) void contextK(const float* __restrict__ X,
                                                const float* __restrict__ w,
                                                float* __restrict__ ctx) {
    const int b     = blockIdx.x;
    const int chunk = blockIdx.y;           // 0..15, each covers 256 timesteps
    const int f     = threadIdx.x;          // 0..255

    const float* xb = X + ((size_t)b * TIME + (size_t)chunk * 256) * F_DIM;
    const float* wb = w + (size_t)b * TIME + chunk * 256;

    float acc = 0.0f;
    for (int t = 0; t < 256; ++t) {
        if ((t & 15) == 0 && (t + 16) < 256)
            __builtin_prefetch(xb + (size_t)(t + 16) * F_DIM + f, 0, 0);
        acc += wb[t] * xb[(size_t)t * F_DIM + f];
    }
    atomicAdd(&ctx[b * F_DIM + f], acc);
}

// ---------------------------------------------------------------------------
// Host launcher
// d_in: inputs(B,T,F) f32 | Wq(H,F) | bq(H) | Wk(H,F) | bk(H) | Ws(1,H) | bs(1)
// d_out: context (B*F) then weights (B*T), f32
// d_ws: 128 KB used for the pre-swizzled bf16 combined weight fragments
// ---------------------------------------------------------------------------
extern "C" void kernel_launch(void* const* d_in, const int* in_sizes, int n_in,
                              void* d_out, int out_size, void* d_ws, size_t ws_size,
                              hipStream_t stream) {
    const float* X  = (const float*)d_in[0];
    const float* Wq = (const float*)d_in[1];
    const float* bq = (const float*)d_in[2];
    const float* Wk = (const float*)d_in[3];
    const float* bk = (const float*)d_in[4];
    const float* Ws = (const float*)d_in[5];
    const float* bs = (const float*)d_in[6];

    float* ctx = (float*)d_out;                       // BATCH*F_DIM
    float* wts = (float*)d_out + BATCH * F_DIM;       // BATCH*TIME
    unsigned short* wcf = (unsigned short*)d_ws;      // 65536 bf16 = 128 KB

    // 1) Fold Wq+Wk -> bf16 B-fragments; zero context output.
    prepK<<<256, 256, 0, stream>>>(Wq, Wk, wcf, ctx);

    // 2) Fused GEMM + tanh + Ws-dot: 262144 rows / 16 per wave / 8 waves per block.
    scoresK<<<M_ROWS / 16 / 8, 256, 0, stream>>>(X, wcf, bq, bk, Ws, bs, wts);

    // 3) Softmax over T per batch row (in place in d_out weights section).
    softmaxK<<<BATCH, 256, 0, stream>>>(wts);

    // 4) Weighted sum of inputs -> context.
    contextK<<<dim3(BATCH, 16), 256, 0, stream>>>(X, wts, ctx);
}
